// SogCLR_Loss_36378372997154
// MI455X (gfx1250) — compile-verified, matrix operand order checked by reading, not run on hardware
//
#include <hip/hip_runtime.h>

typedef __attribute__((ext_vector_type(16))) _Float16 v16h;
typedef __attribute__((ext_vector_type(8)))  _Float16 v8h;
typedef __attribute__((ext_vector_type(8)))  float    v8f;

#define HID   768
#define NROW  4096
#define BHALF 2048
#define TILE  128
#define KC    64
#define NCHUNK (HID / KC)   // 12
#define LSTR  72   // padded LDS row stride (f16 units): 144B = 36 banks, conflict-free b128 reads

// CDNA5 async global->LDS copy path (bypasses VGPR staging; tracked by ASYNCcnt).
// Probe-verified builtin signature:
//   __builtin_amdgcn_global_load_async_to_lds_b128(AS1 int4* src, AS3 int4* dst, imm offset, imm cpol)
#if defined(__gfx1250__) && __has_builtin(__builtin_amdgcn_global_load_async_to_lds_b128) && \
    __has_builtin(__builtin_amdgcn_s_wait_asynccnt)
#define USE_ASYNC_LDS 1
typedef __attribute__((ext_vector_type(4))) int i4;
typedef __attribute__((address_space(1))) i4   gi4;   // global-AS int4
typedef __attribute__((address_space(3))) i4   li4;   // LDS-AS int4
#else
#define USE_ASYNC_LDS 0
#endif

// Scheduling pin: per k-step, group all DS reads ahead of the WMMA burst so the
// waitcnt inserter can emit partial s_wait_dscnt (overlap LDS latency w/ WMMA).
#if __has_builtin(__builtin_amdgcn_sched_group_barrier)
#define SCHED_PIN_KSTEP()                                      \
    do {                                                       \
        __builtin_amdgcn_sched_group_barrier(0x100, 18, 0);    \
        __builtin_amdgcn_sched_group_barrier(0x008,  8, 0);    \
    } while (0)
#else
#define SCHED_PIN_KSTEP() do {} while (0)
#endif

// ---------------------------------------------------------------------------
// init: zero the per-row accumulators and the loss cell
// ---------------------------------------------------------------------------
__global__ void k_init(float* __restrict__ S, float* __restrict__ P, float* __restrict__ out0) {
    int i = blockIdx.x * blockDim.x + threadIdx.x;
    if (i < NROW) { S[i] = 0.0f; P[i] = 0.0f; }
    if (i == 0) out0[0] = 0.0f;
}

// ---------------------------------------------------------------------------
// normalize rows of hidden -> f16 (one 256-thread block per row)
// ---------------------------------------------------------------------------
__global__ void k_norm(const float* __restrict__ hidden, _Float16* __restrict__ Hn) {
    int row = blockIdx.x;
    int tid = threadIdx.x;
    const float* hrow = hidden + (size_t)row * HID;
    float ss = 0.0f;
    for (int c = tid; c < HID; c += 256) { float v = hrow[c]; ss += v * v; }
    for (int m = 16; m; m >>= 1) ss += __shfl_xor(ss, m, 32);
    __shared__ float red[8];
    __shared__ float s_scale;
    if ((tid & 31) == 0) red[tid >> 5] = ss;
    __syncthreads();
    if (tid == 0) {
        float tot = 0.0f;
        for (int w = 0; w < 8; w++) tot += red[w];
        s_scale = 1.0f / fmaxf(sqrtf(tot), 1e-12f);
    }
    __syncthreads();
    float sc = s_scale;
    _Float16* orow = Hn + (size_t)row * HID;
    for (int c = tid; c < HID; c += 256) orow[c] = (_Float16)(hrow[c] * sc);
}

// ---------------------------------------------------------------------------
// diag[i] = dot(Hn[i], Hn[i+2048])  (label logit logits_ab[i,i])
// ---------------------------------------------------------------------------
__global__ void k_diag(const _Float16* __restrict__ Hn, float* __restrict__ diag) {
    int i = blockIdx.x;
    int tid = threadIdx.x;
    const _Float16* a = Hn + (size_t)i * HID;
    const _Float16* b = Hn + (size_t)(i + BHALF) * HID;
    float s = 0.0f;
    for (int c = tid; c < HID; c += 256) s += (float)a[c] * (float)b[c];
    for (int m = 16; m; m >>= 1) s += __shfl_xor(s, m, 32);
    __shared__ float red[8];
    if ((tid & 31) == 0) red[tid >> 5] = s;
    __syncthreads();
    if (tid == 0) {
        float tot = 0.0f;
        for (int w = 0; w < 8; w++) tot += red[w];
        diag[i] = tot;
    }
}

// ---------------------------------------------------------------------------
// A/B fragment load from LDS per the CDNA5 16-bit 16x32 layout:
// lanes 0-15 hold row (lane&15), K {kk..kk+7, kk+16..kk+23};
// lanes 16-31 hold the same rows, K {kk+8..kk+15, kk+24..kk+31}.
// ---------------------------------------------------------------------------
__device__ __forceinline__ v16h load_frag(const _Float16* tile, int lane, int kk) {
    int r   = lane & 15;
    int off = (lane & 16) ? 8 : 0;
    const _Float16* p = tile + r * LSTR + kk + off;
    v8h lo = *(const v8h*)p;
    v8h hi = *(const v8h*)(p + 16);
    return __builtin_shufflevector(lo, hi, 0,1,2,3,4,5,6,7,8,9,10,11,12,13,14,15);
}

// ---------------------------------------------------------------------------
// Gram GEMM + fused exp epilogue.
// Block = 256 threads (8 waves), computes a 128x128 tile of G = Hn*Hn^T.
// Wave w owns rows [w*16, w*16+16) x 128 cols = 8 WMMA accumulators.
// Pipeline: LDS panels are double-buffered; async global->LDS loads for chunk
// c+1 are issued before computing chunk c (hidden behind a full K-chunk of
// WMMAs, resolved by s_wait_asynccnt). Per k-step, sched_group_barrier pins
// all 18 ds_load_b128 ahead of the 8-WMMA burst for partial dscnt waits.
// Epilogue: skip j==r and j==r^2048, accumulate S[r]+=e^{10G}, P[r]+=e^{10G}*G.
// ---------------------------------------------------------------------------
__global__ __launch_bounds__(256)
void k_gram(const _Float16* __restrict__ Hn, float* __restrict__ S, float* __restrict__ P) {
    __shared__ _Float16 lA[2][TILE * LSTR];
    __shared__ _Float16 lB[2][TILE * LSTR];

    int tid  = threadIdx.x;
    int wave = tid >> 5;
    int lane = tid & 31;
    int rowBase = blockIdx.y * TILE;
    int colBase = blockIdx.x * TILE;

    // stage one 128x64 A panel + one 128x64 B panel into LDS buffer `buf`
    auto stage = [&](int buf, int k0) {
        for (int i = tid; i < TILE * (KC / 8); i += 256) {
            int r = i >> 3;
            int c = (i & 7) * 8;
#if USE_ASYNC_LDS
            __builtin_amdgcn_global_load_async_to_lds_b128(
                (gi4*)&Hn[(size_t)(rowBase + r) * HID + k0 + c],
                (li4*)&lA[buf][r * LSTR + c], 0, 0);
            __builtin_amdgcn_global_load_async_to_lds_b128(
                (gi4*)&Hn[(size_t)(colBase + r) * HID + k0 + c],
                (li4*)&lB[buf][r * LSTR + c], 0, 0);
#else
            *(v8h*)&lA[buf][r * LSTR + c] = *(const v8h*)&Hn[(size_t)(rowBase + r) * HID + k0 + c];
            *(v8h*)&lB[buf][r * LSTR + c] = *(const v8h*)&Hn[(size_t)(colBase + r) * HID + k0 + c];
#endif
        }
    };

    v8f acc[8] = {};

    stage(0, 0);   // preamble: chunk 0 in flight

    for (int c0 = 0; c0 < NCHUNK; c0++) {
        int cur = c0 & 1;
#if USE_ASYNC_LDS
        __builtin_amdgcn_s_wait_asynccnt(0);   // my chunk-c0 loads have landed
#endif
        __syncthreads();                        // everyone's chunk-c0 data visible
        if (c0 + 1 < NCHUNK) stage(1 - cur, (c0 + 1) * KC);   // prefetch next chunk

        const _Float16* atile = &lA[cur][(wave * 16) * LSTR];
        const _Float16* btile = &lB[cur][0];
#pragma unroll
        for (int kk = 0; kk < KC; kk += 32) {
            v16h a = load_frag(atile, lane, kk);
            v16h b[8];
#pragma unroll
            for (int t = 0; t < 8; t++) b[t] = load_frag(btile + t * 16 * LSTR, lane, kk);
#pragma unroll
            for (int t = 0; t < 8; t++) {
                acc[t] = __builtin_amdgcn_wmma_f32_16x16x32_f16(
                    false, a, false, b[t], (short)0, acc[t], false, false);
            }
            SCHED_PIN_KSTEP();   // 18 DS reads, then 8 WMMAs
        }
        __syncthreads();   // buf[cur] free for the prefetch two chunks ahead
    }

    // fused epilogue: C layout -> M = e + 8*(lane>=16), N = lane&15
    int half = lane >> 4;
    int lc   = lane & 15;
#pragma unroll
    for (int e = 0; e < 8; e++) {
        int gm = rowBase + wave * 16 + half * 8 + e;
        float s = 0.0f, p = 0.0f;
#pragma unroll
        for (int t = 0; t < 8; t++) {
            int gn = colBase + t * 16 + lc;
            float v = acc[t][e];
            if ((gn != gm) && (gn != (gm ^ BHALF))) {
                float ex = __expf(v * 10.0f);   // 1/T = 10
                s += ex;
                p += ex * v;
            }
        }
        // reduce across the 16 lanes of this row half (xor masks stay in-half)
        s += __shfl_xor(s, 1, 32);  p += __shfl_xor(p, 1, 32);
        s += __shfl_xor(s, 2, 32);  p += __shfl_xor(p, 2, 32);
        s += __shfl_xor(s, 4, 32);  p += __shfl_xor(p, 4, 32);
        s += __shfl_xor(s, 8, 32);  p += __shfl_xor(p, 8, 32);
        if (lc == 0) {
            atomicAdd(&S[gm], s);
            atomicAdd(&P[gm], p);
        }
    }
}

// ---------------------------------------------------------------------------
// finalize: u1/u2, scatter u_new, per-row sce, reduce to loss
// ---------------------------------------------------------------------------
__global__ void k_final(const float* __restrict__ S, const float* __restrict__ P,
                        const float* __restrict__ diag, const float* __restrict__ u,
                        const long long* __restrict__ index, float* __restrict__ out) {
    int i = blockIdx.x * 256 + threadIdx.x;
    float sce = 0.0f;
    if (i < BHALF) {
        long long idx = index[i];
        float u_old = u[idx];
        const float denom = 2.0f * (float)(BHALF - 1);   // 4094
        float u1 = 0.2f * u_old + 0.8f * S[i] / denom;
        float u2 = 0.2f * u_old + 0.8f * S[i + BHALF] / denom;
        out[1 + idx] = u1 + u2;
        float d = diag[i];
        sce = -(d - P[i] / (u1 * denom)) - (d - P[i + BHALF] / (u2 * denom));
    }
    for (int m = 16; m; m >>= 1) sce += __shfl_xor(sce, m, 32);
    __shared__ float red[8];
    if ((threadIdx.x & 31) == 0) red[threadIdx.x >> 5] = sce;
    __syncthreads();
    if (threadIdx.x == 0) {
        float tot = 0.0f;
        for (int w = 0; w < 8; w++) tot += red[w];
        atomicAdd(out, tot * (1.0f / (float)BHALF));
    }
}

// ---------------------------------------------------------------------------
extern "C" void kernel_launch(void* const* d_in, const int* in_sizes, int n_in,
                              void* d_out, int out_size, void* d_ws, size_t ws_size,
                              hipStream_t stream) {
    const float*     hidden = (const float*)d_in[0];
    const float*     u      = (const float*)d_in[1];
    const long long* index  = (const long long*)d_in[2];   // int64 indices
    float*           out    = (float*)d_out;               // [loss, u_new(15M)]
    int n_u = in_sizes[1];

    // workspace layout
    _Float16* Hn   = (_Float16*)d_ws;                                   // 4096*768 f16 (6.3MB)
    float*    S    = (float*)((char*)d_ws + (size_t)NROW * HID * sizeof(_Float16));
    float*    P    = S + NROW;
    float*    diag = P + NROW;

    k_init<<<16, 256, 0, stream>>>(S, P, out);
    k_norm<<<NROW, 256, 0, stream>>>(hidden, Hn);
    k_diag<<<BHALF, 256, 0, stream>>>(Hn, diag);
    k_gram<<<dim3(NROW / TILE, NROW / TILE), 256, 0, stream>>>(Hn, S, P);
    // pass-through of u into the output (dominant byte traffic: ~120MB r+w, ~5us @ 23.3TB/s)
    (void)hipMemcpyAsync(out + 1, u, (size_t)n_u * sizeof(float), hipMemcpyDeviceToDevice, stream);
    k_final<<<(BHALF + 255) / 256, 256, 0, stream>>>(S, P, diag, u, index, out);
}
